// EnhancedGNN_75179107549620
// MI455X (gfx1250) — matrix operand order repaired and checked by reading, not compiled, not run on hardware
//
#include <hip/hip_runtime.h>
#include <math.h>

typedef _Float16 half_t;
typedef __attribute__((ext_vector_type(16))) _Float16 v16h;
typedef __attribute__((ext_vector_type(8)))  _Float16 v8h;
typedef __attribute__((ext_vector_type(4)))  _Float16 v4h;
typedef __attribute__((ext_vector_type(8)))  float    v8f;
typedef __attribute__((ext_vector_type(4)))  int      v4i;

#define KDIM 128
#define BN_EPS 1e-5f

// CDNA5 async direct-to-LDS staging (bypasses VGPRs, tracked by ASYNCcnt).
// Gated on builtin availability so the code compiles on both toolchains.
#if __has_builtin(__builtin_amdgcn_global_load_async_to_lds_b128) && \
    __has_builtin(__builtin_amdgcn_s_wait_asynccnt)
#define GNN_ASYNC_LDS 1
typedef __attribute__((address_space(1))) v4i* gnn_gptr_v4i;
typedef __attribute__((address_space(3))) v4i* gnn_lptr_v4i;
#endif

// ---------------------------------------------------------------------------
// zero kernel
// ---------------------------------------------------------------------------
__global__ void gnn_zero_f32(float* __restrict__ p, int n) {
    int i = blockIdx.x * blockDim.x + threadIdx.x;
    if (i < n) p[i] = 0.0f;
}

// ---------------------------------------------------------------------------
// weight transpose + f32->f16 convert:  W[K][Nsrc] -> Wt[Npad][K] (zero pad)
// ---------------------------------------------------------------------------
__global__ void gnn_wtrans(const float* __restrict__ W, half_t* __restrict__ Wt,
                           int Nsrc, int Npad) {
    int idx = blockIdx.x * blockDim.x + threadIdx.x;
    if (idx >= Npad * KDIM) return;
    int n = idx / KDIM, k = idx - n * KDIM;
    float v = (n < Nsrc) ? W[(size_t)k * Nsrc + n] : 0.0f;
    Wt[idx] = (half_t)v;
}

// ---------------------------------------------------------------------------
// degree: deg[dst] += 1
// ---------------------------------------------------------------------------
__global__ void gnn_degree(const int* __restrict__ edst, float* __restrict__ deg, int E) {
    int i = blockIdx.x * blockDim.x + threadIdx.x;
    if (i < E) unsafeAtomicAdd(deg + edst[i], 1.0f);
}

// ---------------------------------------------------------------------------
// scatter-add of 128-wide rows: one wave per edge, float4 per lane
// ---------------------------------------------------------------------------
__global__ void gnn_scatter_add(const float* __restrict__ feat,
                                const int* __restrict__ esrc,
                                const int* __restrict__ edst,
                                float* __restrict__ agg, int E) {
    int gtid = blockIdx.x * blockDim.x + threadIdx.x;
    int e = gtid >> 5;
    int lane = gtid & 31;
    if (e >= E) return;
    int s = esrc[e], d = edst[e];
    float4 v = ((const float4*)(feat + (size_t)s * KDIM))[lane];
    float* dp = agg + (size_t)d * KDIM + lane * 4;
    unsafeAtomicAdd(dp + 0, v.x);
    unsafeAtomicAdd(dp + 1, v.y);
    unsafeAtomicAdd(dp + 2, v.z);
    unsafeAtomicAdd(dp + 3, v.w);
}

// ---------------------------------------------------------------------------
// convert to f16 with optional add (b) and optional mean scale (deg)
//   out16 = (h16)( (a [+ b]) [* 1/max(deg,1)] )       4 elems / thread
// ---------------------------------------------------------------------------
__global__ void gnn_cvt16(const float* __restrict__ a, const float* __restrict__ b,
                          const float* __restrict__ deg, half_t* __restrict__ o,
                          int n_rows) {
    int idx = blockIdx.x * blockDim.x + threadIdx.x;   // one per float4
    if (idx >= n_rows * (KDIM / 4)) return;
    int r = idx >> 5;
    float s = (deg != nullptr) ? (1.0f / fmaxf(deg[r], 1.0f)) : 1.0f;
    float4 va = ((const float4*)a)[idx];
    if (b != nullptr) {
        float4 vb = ((const float4*)b)[idx];
        va.x += vb.x; va.y += vb.y; va.z += vb.z; va.w += vb.w;
    }
    v4h h;
    h.x = (half_t)(va.x * s); h.y = (half_t)(va.y * s);
    h.z = (half_t)(va.z * s); h.w = (half_t)(va.w * s);
    ((v4h*)o)[idx] = h;
}

// ---------------------------------------------------------------------------
// WMMA GEMM:  out = A1 @ Wt1^T (+ A2 @ Wt2^T) + bias
//   A*  : [n_rows][128] f16 row-major
//   Wt* : [NT*16][128] f16  (transposed weights, K contiguous, zero padded)
//   MODE 0: outF32[n_rows][NT*16] = pre-activation (for BN)
//   MODE 1: outF16[n_rows][NT*16] = relu(pre)       (GIN mid layer)
// 256 threads = 8 waves; block tile = 128 rows x NT*16 cols; K loop 4 x 32.
// ---------------------------------------------------------------------------
template <int NT, int NMAT, int MODE>
__global__ __launch_bounds__(256)
void gnn_gemm_wmma(const half_t* __restrict__ A1, const half_t* __restrict__ A2,
                   const half_t* __restrict__ Wt1, const half_t* __restrict__ Wt2,
                   const float* __restrict__ bias,
                   float* __restrict__ outF32, half_t* __restrict__ outF16,
                   int n_rows, int n_valid) {
    __shared__ half_t sW[NMAT][NT * 16][KDIM];

    const int tid = threadIdx.x;
    constexpr int NV = NT * 16 * KDIM / 8;   // 16-byte vectors per matrix

#if defined(GNN_ASYNC_LDS)
    // stage weights via GLOBAL_LOAD_ASYNC_TO_LDS_B128 (no VGPR round trip)
    #pragma unroll
    for (int m = 0; m < NMAT; ++m) {
        char* gsrc = (char*)((m == 0) ? Wt1 : Wt2);
        char* lds  = (char*)&sW[m][0][0];
        #pragma unroll
        for (int v = 0; v < NV / 256; ++v) {
            const int off = (tid + v * 256) * 16;
            __builtin_amdgcn_global_load_async_to_lds_b128(
                (gnn_gptr_v4i)(gsrc + off),
                (gnn_lptr_v4i)(lds + off),
                0, 0);
        }
    }
    __builtin_amdgcn_s_wait_asynccnt(0);
#else
    // fallback: plain global->VGPR->LDS copy (uniform trip count, no divergence)
    for (int m = 0; m < NMAT; ++m) {
        const v8h* src = (const v8h*)((m == 0) ? Wt1 : Wt2);
        v8h* dst = (v8h*)&sW[m][0][0];
        #pragma unroll
        for (int v = 0; v < NV / 256; ++v) dst[tid + v * 256] = src[tid + v * 256];
    }
#endif
    __syncthreads();

    const int wave  = tid >> 5;
    const int lane  = tid & 31;
    const int lhalf = lane & 15;
    const int hsel  = lane >> 4;          // 0: lanes 0-15, 1: lanes 16-31

    const int rowBase = blockIdx.x * 128 + wave * 16;
    int arow = rowBase + lhalf;
    if (arow > n_rows - 1) arow = n_rows - 1;       // clamp OOB A reads (store is guarded)

    const half_t* a1row = A1 + (size_t)arow * KDIM;
    const half_t* a2row = (NMAT == 2) ? (A2 + (size_t)arow * KDIM) : a1row;

    v8f acc[NT];
    #pragma unroll
    for (int t = 0; t < NT; ++t)
        #pragma unroll
        for (int i = 0; i < 8; ++i) acc[t][i] = 0.0f;

    #pragma unroll
    for (int kc = 0; kc < KDIM; kc += 32) {
        // A fragment (16-bit 16x32 layout: lanes 0-15 -> K 0..7 & 16..23, lanes 16-31 -> +8)
        union { v16h v; v8h h[2]; } a1, a2;
        const int aoff = kc + hsel * 8;
        a1.h[0] = *(const v8h*)(a1row + aoff);
        a1.h[1] = *(const v8h*)(a1row + aoff + 16);
        if (NMAT == 2) {
            a2.h[0] = *(const v8h*)(a2row + aoff);
            a2.h[1] = *(const v8h*)(a2row + aoff + 16);
        }
        #pragma unroll
        for (int t = 0; t < NT; ++t) {
            // B fragment (32x16): lane = column n, K 0..7 & 16..23; lanes 16-31 = +8
            union { v16h v; v8h h[2]; } b;
            const half_t* bp = &sW[0][t * 16 + lhalf][kc + hsel * 8];
            b.h[0] = *(const v8h*)bp;
            b.h[1] = *(const v8h*)(bp + 16);
            acc[t] = __builtin_amdgcn_wmma_f32_16x16x32_f16(
                false, a1.v, false, b.v, (short)0, acc[t], false, false);
            if (NMAT == 2) {
                const half_t* bp2 = &sW[1][t * 16 + lhalf][kc + hsel * 8];
                b.h[0] = *(const v8h*)bp2;
                b.h[1] = *(const v8h*)(bp2 + 16);
                acc[t] = __builtin_amdgcn_wmma_f32_16x16x32_f16(
                    false, a2.v, false, b.v, (short)0, acc[t], false, false);
            }
        }
    }

    // epilogue: C/D layout — lane gives column, VGPR r gives row (+8 for lanes 16-31)
    #pragma unroll
    for (int t = 0; t < NT; ++t) {
        const int n = t * 16 + lhalf;
        float bv = 0.0f;
        if (bias != nullptr && n < n_valid) bv = bias[n];
        #pragma unroll
        for (int r = 0; r < 8; ++r) {
            const int grow = rowBase + hsel * 8 + r;
            if (grow < n_rows && n < n_valid) {
                float v = acc[t][r] + bv;
                if (MODE == 0) {
                    outF32[(size_t)grow * (NT * 16) + n] = v;
                } else {
                    outF16[(size_t)grow * (NT * 16) + n] = (half_t)fmaxf(v, 0.0f);
                }
            }
        }
    }
}

// ---------------------------------------------------------------------------
// BN stats: per-channel sum & sumsq (stats[0..127]=sum, stats[128..255]=sumsq)
// ---------------------------------------------------------------------------
#define BN_ROWS 512
__global__ __launch_bounds__(KDIM)
void gnn_bn_stats(const float* __restrict__ pre, float* __restrict__ stats, int n_rows) {
    int c = threadIdx.x;
    int r0 = blockIdx.x * BN_ROWS;
    int r1 = r0 + BN_ROWS; if (r1 > n_rows) r1 = n_rows;
    float s = 0.0f, s2 = 0.0f;
    for (int r = r0; r < r1; ++r) {
        float v = pre[(size_t)r * KDIM + c];
        s += v; s2 += v * v;
    }
    unsafeAtomicAdd(&stats[c], s);
    unsafeAtomicAdd(&stats[KDIM + c], s2);
}

// ---------------------------------------------------------------------------
// BN apply (+ReLU, + optional residual), writes f32 h and f16 mirror
// ---------------------------------------------------------------------------
__global__ void gnn_bn_apply(const float* __restrict__ pre, const float* __restrict__ stats,
                             const float* __restrict__ gamma, const float* __restrict__ beta,
                             const float* __restrict__ res,
                             float* __restrict__ h, half_t* __restrict__ h16, int n_rows) {
    int idx = blockIdx.x * blockDim.x + threadIdx.x;
    if (idx >= n_rows * KDIM) return;
    int c = idx & (KDIM - 1);
    float inv_n = 1.0f / (float)n_rows;
    float mean = stats[c] * inv_n;
    float var  = stats[KDIM + c] * inv_n - mean * mean;
    float v = (pre[idx] - mean) * rsqrtf(var + BN_EPS) * gamma[c] + beta[c];
    v = fmaxf(v, 0.0f);
    if (res != nullptr) v += res[idx];
    h[idx] = v;
    h16[idx] = (half_t)v;
}

// ---------------------------------------------------------------------------
// log_softmax over 40 valid columns of a 48-stride buffer; one wave per row
// ---------------------------------------------------------------------------
__global__ void gnn_log_softmax(const float* __restrict__ pre48, float* __restrict__ out,
                                int n_rows) {
    int gtid = blockIdx.x * blockDim.x + threadIdx.x;
    int row = gtid >> 5;
    int lane = gtid & 31;
    if (row >= n_rows) return;
    const float* p = pre48 + (size_t)row * 48;
    float v0 = (lane < 40) ? p[lane] : -INFINITY;
    float v1 = (lane + 32 < 40) ? p[lane + 32] : -INFINITY;
    float m = fmaxf(v0, v1);
    #pragma unroll
    for (int o = 16; o > 0; o >>= 1) m = fmaxf(m, __shfl_xor(m, o, 32));
    float e = ((lane < 40) ? __expf(v0 - m) : 0.0f) +
              ((lane + 32 < 40) ? __expf(v1 - m) : 0.0f);
    #pragma unroll
    for (int o = 16; o > 0; o >>= 1) e += __shfl_xor(e, o, 32);
    float lse = m + __logf(e);
    if (lane < 40)      out[(size_t)row * 40 + lane]      = v0 - lse;
    if (lane + 32 < 40) out[(size_t)row * 40 + lane + 32] = v1 - lse;
}

// ---------------------------------------------------------------------------
// host orchestration
// ---------------------------------------------------------------------------
extern "C" void kernel_launch(void* const* d_in, const int* in_sizes, int n_in,
                              void* d_out, int out_size, void* d_ws, size_t ws_size,
                              hipStream_t stream) {
    const int N = in_sizes[0] / KDIM;
    const int E = in_sizes[1] / 2;

    const float* x     = (const float*)d_in[0];
    const int*   esrc  = (const int*)d_in[1];
    const int*   edst  = esrc + E;
    const float* s0wl  = (const float*)d_in[2];
    const float* s0wr  = (const float*)d_in[3];
    const float* s0b   = (const float*)d_in[4];
    const float* gw1   = (const float*)d_in[5];
    const float* gb1   = (const float*)d_in[6];
    const float* gw2   = (const float*)d_in[7];
    const float* gb2   = (const float*)d_in[8];
    const float* s2wl  = (const float*)d_in[9];
    const float* s2wr  = (const float*)d_in[10];
    const float* s2b   = (const float*)d_in[11];
    const float* fwl   = (const float*)d_in[12];
    const float* fwr   = (const float*)d_in[13];
    const float* fb    = (const float*)d_in[14];
    const float* gam   = (const float*)d_in[15];
    const float* bet   = (const float*)d_in[16];
    float* out = (float*)d_out;

    // workspace carve
    uint8_t* wp = (uint8_t*)d_ws;
    auto take = [&](size_t bytes) -> void* {
        void* r = (void*)wp;
        wp += (bytes + 255) & ~(size_t)255;
        return r;
    };
    const size_t NF = (size_t)N * KDIM;
    float*  AGG  = (float*)take(NF * 4);
    float*  H0   = (float*)take(NF * 4);
    float*  H1   = (float*)take(NF * 4);
    float*  PRE  = (float*)take(NF * 4);
    half_t* F16A = (half_t*)take(NF * 2);   // agg16 / g16
    half_t* F16H = (half_t*)take(NF * 2);   // self/h16
    half_t* F16T = (half_t*)take(NF * 2);   // GIN mid
    half_t* WT_S0L = (half_t*)take(128 * KDIM * 2);
    half_t* WT_S0R = (half_t*)take(128 * KDIM * 2);
    half_t* WT_G1  = (half_t*)take(128 * KDIM * 2);
    half_t* WT_G2  = (half_t*)take(128 * KDIM * 2);
    half_t* WT_S2L = (half_t*)take(128 * KDIM * 2);
    half_t* WT_S2R = (half_t*)take(128 * KDIM * 2);
    half_t* WT_FL  = (half_t*)take(48 * KDIM * 2);
    half_t* WT_FR  = (half_t*)take(48 * KDIM * 2);
    float*  deg   = (float*)take((size_t)N * 4);
    float*  stats = (float*)take(256 * 4);

    const int TB = 256;
    const int elemBlocks  = (int)((NF + TB - 1) / TB);
    const int vec4Blocks  = (int)((NF / 4 + TB - 1) / TB);
    const int gemmBlocks  = (N + 127) / 128;
    const int edgeWaveBlk = (int)(((size_t)E * 32 + TB - 1) / TB);
    const int bnBlocks    = (N + BN_ROWS - 1) / BN_ROWS;
    const int rowWaveBlk  = (int)(((size_t)N * 32 + TB - 1) / TB);

    auto zero = [&](float* ptr, size_t n) {
        gnn_zero_f32<<<(int)((n + TB - 1) / TB), TB, 0, stream>>>(ptr, (int)n);
    };

    // ---- weight prep ----
    gnn_wtrans<<<(128 * KDIM + TB - 1) / TB, TB, 0, stream>>>(s0wl, WT_S0L, 128, 128);
    gnn_wtrans<<<(128 * KDIM + TB - 1) / TB, TB, 0, stream>>>(s0wr, WT_S0R, 128, 128);
    gnn_wtrans<<<(128 * KDIM + TB - 1) / TB, TB, 0, stream>>>(gw1,  WT_G1,  128, 128);
    gnn_wtrans<<<(128 * KDIM + TB - 1) / TB, TB, 0, stream>>>(gw2,  WT_G2,  128, 128);
    gnn_wtrans<<<(128 * KDIM + TB - 1) / TB, TB, 0, stream>>>(s2wl, WT_S2L, 128, 128);
    gnn_wtrans<<<(128 * KDIM + TB - 1) / TB, TB, 0, stream>>>(s2wr, WT_S2R, 128, 128);
    gnn_wtrans<<<(48 * KDIM + TB - 1) / TB, TB, 0, stream>>>(fwl, WT_FL, 40, 48);
    gnn_wtrans<<<(48 * KDIM + TB - 1) / TB, TB, 0, stream>>>(fwr, WT_FR, 40, 48);

    // ---- degree (shared by all mean aggregations) ----
    zero(deg, N);
    gnn_degree<<<(E + TB - 1) / TB, TB, 0, stream>>>(edst, deg, E);

    // ---- layer 0: SAGE(x) -> BN -> ReLU ----
    zero(AGG, NF);
    gnn_scatter_add<<<edgeWaveBlk, TB, 0, stream>>>(x, esrc, edst, AGG, E);
    gnn_cvt16<<<vec4Blocks, TB, 0, stream>>>(AGG, nullptr, deg, F16A, N);  // agg16 = mean
    gnn_cvt16<<<vec4Blocks, TB, 0, stream>>>(x, nullptr, nullptr, F16H, N); // x16
    gnn_gemm_wmma<8, 2, 0><<<gemmBlocks, TB, 0, stream>>>(
        F16A, F16H, WT_S0L, WT_S0R, s0b, PRE, nullptr, N, 128);
    zero(stats, 256);
    gnn_bn_stats<<<bnBlocks, KDIM, 0, stream>>>(PRE, stats, N);
    gnn_bn_apply<<<elemBlocks, TB, 0, stream>>>(PRE, stats, gam + 0, bet + 0,
                                                nullptr, H0, F16H, N);

    // ---- layer 1: GIN: mlp(h + sum_agg(h)) -> BN -> ReLU -> +res ----
    zero(AGG, NF);
    gnn_scatter_add<<<edgeWaveBlk, TB, 0, stream>>>(H0, esrc, edst, AGG, E);
    gnn_cvt16<<<vec4Blocks, TB, 0, stream>>>(AGG, H0, nullptr, F16A, N);   // g16 = agg + h
    gnn_gemm_wmma<8, 1, 1><<<gemmBlocks, TB, 0, stream>>>(
        F16A, nullptr, WT_G1, nullptr, gb1, nullptr, F16T, N, 128);        // t = relu(.)
    gnn_gemm_wmma<8, 1, 0><<<gemmBlocks, TB, 0, stream>>>(
        F16T, nullptr, WT_G2, nullptr, gb2, PRE, nullptr, N, 128);
    zero(stats, 256);
    gnn_bn_stats<<<bnBlocks, KDIM, 0, stream>>>(PRE, stats, N);
    gnn_bn_apply<<<elemBlocks, TB, 0, stream>>>(PRE, stats, gam + 128, bet + 128,
                                                H0, H1, F16H, N);

    // ---- layer 2: SAGE(h) -> BN -> ReLU -> +res ----
    zero(AGG, NF);
    gnn_scatter_add<<<edgeWaveBlk, TB, 0, stream>>>(H1, esrc, edst, AGG, E);
    gnn_cvt16<<<vec4Blocks, TB, 0, stream>>>(AGG, nullptr, deg, F16A, N);
    gnn_gemm_wmma<8, 2, 0><<<gemmBlocks, TB, 0, stream>>>(
        F16A, F16H, WT_S2L, WT_S2R, s2b, PRE, nullptr, N, 128);
    zero(stats, 256);
    gnn_bn_stats<<<bnBlocks, KDIM, 0, stream>>>(PRE, stats, N);
    gnn_bn_apply<<<elemBlocks, TB, 0, stream>>>(PRE, stats, gam + 256, bet + 256,
                                                H1, H0, F16H, N);

    // ---- final: SAGE(h) -> log_softmax ----
    zero(AGG, NF);
    gnn_scatter_add<<<edgeWaveBlk, TB, 0, stream>>>(H0, esrc, edst, AGG, E);
    gnn_cvt16<<<vec4Blocks, TB, 0, stream>>>(AGG, nullptr, deg, F16A, N);
    gnn_gemm_wmma<3, 2, 0><<<gemmBlocks, TB, 0, stream>>>(
        F16A, F16H, WT_FL, WT_FR, fb, PRE, nullptr, N, 40);                // stride 48
    gnn_log_softmax<<<rowWaveBlk, TB, 0, stream>>>(PRE, out, N);
}
